// LSTMModel_35132832481897
// MI455X (gfx1250) — matrix operand order repaired and checked by reading, not compiled, not run on hardware
//
#include <hip/hip_runtime.h>
#include <hip/hip_bf16.h>

// ---------------------------------------------------------------------------
// Fused 2-layer LSTM + dense head for MI455X (gfx1250, wave32, WMMA).
// 64 workgroups x 16 batch rows; per-WG: all weights staged to LDS as bf16,
// 512 sequential timesteps with v_wmma_f32_16x16x32_bf16, cell state in VGPRs.
// ---------------------------------------------------------------------------

typedef __attribute__((ext_vector_type(16))) __bf16 v16bf;
typedef __attribute__((ext_vector_type(8)))  __bf16 v8bf;
typedef __attribute__((ext_vector_type(8)))  float  v8f;

#define TPB   256            // 8 waves of 32
#define BATCH 1024
#define TSTEPS 512
#define FIN   9
#define U1    80
#define U2    128
#define G1    (4*U1)         // 320
#define G2    (4*U2)         // 512
#define K1P   96             // 80 (h1) + 9 (x) padded to 3*32
#define K2P   224            // 80 (h1) + 128 (h2) padded to 7*32
#define BT    16             // batch tile per workgroup

// LDS layout (elements)
#define UC1_ELEMS (G1*K1P)       // 30720  bf16
#define UC2_ELEMS (G2*K2P)       // 114688 bf16
#define H1CAT_ELEMS (BT*K1P)     // 1536   bf16
#define H2CAT_ELEMS (BT*K2P)     // 3584   bf16
#define DB0_ELEMS (BT*U2)        // 2048   f32
#define DB1_ELEMS (BT*U1)        // 1280   f32
#define SMEM_BYTES ((UC1_ELEMS+UC2_ELEMS+H1CAT_ELEMS+H2CAT_ELEMS)*2 + (DB0_ELEMS+DB1_ELEMS)*4)

__device__ __forceinline__ __bf16 f2bf(float f) {
    unsigned u = __builtin_bit_cast(unsigned, f);
    u += 0x7FFFu + ((u >> 16) & 1u);                 // round-to-nearest-even
    unsigned short h = (unsigned short)(u >> 16);
    return __builtin_bit_cast(__bf16, h);
}

__device__ __forceinline__ float sigm(float x) { return 1.0f / (1.0f + __expf(-x)); }

__device__ __forceinline__ v8f wmma_bf16(v16bf a, v16bf b, v8f c) {
    // D = A(16x32 bf16) * B(32x16 bf16) + C(16x16 f32)
    return __builtin_amdgcn_wmma_f32_16x16x32_bf16(
        /*neg_a=*/false, a, /*neg_b=*/false, b,
        /*c_mod=*/(short)0, c, /*reuse_a=*/false, /*reuse_b=*/false);
}

// A-operand assembly (contiguous K order, matching B):
// lane<16: M=lane, K = base+[0..15]
// lane>=16: M=lane-16, K = base+16+[0..15]
__device__ __forceinline__ v16bf load_a(const __bf16* row, int kk, int hi) {
    return *(const v16bf*)(row + kk*32 + hi*16);
}

__device__ __forceinline__ void dense_layer(const float* in, float* out,
                                            const float* W, const float* b,
                                            int din, int dout, bool relu_act, int tid) {
    for (int idx = tid; idx < BT*dout; idx += TPB) {
        int m = idx / dout, o = idx - m*dout;
        float s = b[o];
        for (int k = 0; k < din; ++k) s = fmaf(in[m*din + k], W[k*dout + o], s);
        out[m*dout + o] = relu_act ? fmaxf(s, 0.0f) : sigm(s);
    }
}

__global__ __launch_bounds__(TPB, 1)
void lstm_fused_kernel(const float* __restrict__ xg,
                       const float* __restrict__ W1,  const float* __restrict__ Ur1, const float* __restrict__ b1,
                       const float* __restrict__ W2,  const float* __restrict__ Ur2, const float* __restrict__ b2,
                       const float* __restrict__ Wd1, const float* __restrict__ bd1,
                       const float* __restrict__ Wd2, const float* __restrict__ bd2,
                       const float* __restrict__ Wd3, const float* __restrict__ bd3,
                       const float* __restrict__ Wd4, const float* __restrict__ bd4,
                       const float* __restrict__ Wd5, const float* __restrict__ bd5,
                       float* __restrict__ out) {
    extern __shared__ __align__(32) char smem_raw[];
    __bf16* Uc1T  = (__bf16*)smem_raw;               // [G1][K1P] col-major-per-output
    __bf16* Uc2T  = Uc1T + UC1_ELEMS;                // [G2][K2P]
    __bf16* h1cat = Uc2T + UC2_ELEMS;                // [BT][K1P] : h1 | x_t | 0-pad
    __bf16* h2cat = h1cat + H1CAT_ELEMS;             // [BT][K2P] : h1_t | h2 | 0-pad
    float*  dbuf0 = (float*)(h2cat + H2CAT_ELEMS);   // [BT][128]
    float*  dbuf1 = dbuf0 + DB0_ELEMS;               // [BT][80]

    const int tid  = threadIdx.x;
    const int wave = tid >> 5;
    const int lane = tid & 31;
    const int n16  = lane & 15;
    const int hi   = lane >> 4;
    const int b0   = blockIdx.x * BT;

    // ---- Stage combined weights to LDS as bf16 (column-major: [col][k]) ----
    for (int i = tid; i < UC1_ELEMS; i += TPB) {
        int col = i / K1P, k = i - col*K1P;
        float v = 0.0f;
        if (k < U1)              v = Ur1[k*G1 + col];          // recurrent rows
        else if (k < U1 + FIN)   v = W1[(k - U1)*G1 + col];    // input rows
        Uc1T[i] = f2bf(v);
    }
    for (int i = tid; i < UC2_ELEMS; i += TPB) {
        int col = i / K2P, k = i - col*K2P;
        float v = 0.0f;
        if (k < U1)              v = W2[k*G2 + col];           // h1 -> layer2 rows
        else if (k < U1 + U2)    v = Ur2[(k - U1)*G2 + col];   // recurrent rows
        Uc2T[i] = f2bf(v);
    }
    const __bf16 bz = __builtin_bit_cast(__bf16, (unsigned short)0);
    for (int i = tid; i < H1CAT_ELEMS; i += TPB) h1cat[i] = bz;
    for (int i = tid; i < H2CAT_ELEMS; i += TPB) h2cat[i] = bz;

    // Per-wave gate biases (column depends only on lane within the N-tile)
    float bia1[4] = {0, 0, 0, 0}, bia2[4];
    if (wave < 5) {
        #pragma unroll
        for (int q = 0; q < 4; ++q) bia1[q] = b1[q*U1 + wave*16 + n16];
    }
    #pragma unroll
    for (int q = 0; q < 4; ++q) bia2[q] = b2[q*U2 + wave*16 + n16];

    v8f c1 = {}, c2 = {};          // cell state lives in VGPRs for all 512 steps
    float h2v[8] = {0,0,0,0,0,0,0,0};

    __syncthreads();

    // ======================== time recurrence ========================
    for (int t = 0; t < TSTEPS; ++t) {
        // stage x_t (16 rows x 9 feats) into h1cat columns [80..88]
        if (tid < BT*FIN) {
            int r = tid / FIN, f = tid - r*FIN;
            const float* xp = xg + ((size_t)(b0 + r) * TSTEPS + t) * FIN + f;
            h1cat[r*K1P + U1 + f] = f2bf(*xp);
            if (t + 1 < TSTEPS) __builtin_prefetch(xp + FIN, 0, 0);
        }
        __syncthreads();                       // (1) x_t + h2_{t-1} visible

        // ---- layer 1: z1 = [h1|x] @ Uc1 ; waves 0..4 own gate quadruples ----
        float h1v[8];
        if (wave < 5) {
            const __bf16* arow = h1cat + n16*K1P;
            v16bf a0 = load_a(arow, 0, hi);
            v16bf a1 = load_a(arow, 1, hi);
            v16bf a2 = load_a(arow, 2, hi);
            v8f z1[4];
            #pragma unroll
            for (int q = 0; q < 4; ++q) {
                const __bf16* bcol = Uc1T + (q*U1 + wave*16 + n16)*K1P + hi*16;
                v8f acc = {};
                acc = wmma_bf16(a0, *(const v16bf*)(bcol),      acc);
                acc = wmma_bf16(a1, *(const v16bf*)(bcol + 32), acc);
                acc = wmma_bf16(a2, *(const v16bf*)(bcol + 64), acc);
                z1[q] = acc;
            }
            #pragma unroll
            for (int r = 0; r < 8; ++r) {       // gate order i,f,g,o ; act=tanh
                float iv = sigm(z1[0][r] + bia1[0]);
                float fv = sigm(z1[1][r] + bia1[1]);
                float gv = tanhf(z1[2][r] + bia1[2]);
                float ov = sigm(z1[3][r] + bia1[3]);
                float c  = fv*c1[r] + iv*gv;
                c1[r] = c;
                h1v[r] = ov * tanhf(c);
            }
        }
        __syncthreads();                       // (2) all layer1 A-reads done

        if (wave < 5) {                         // publish h1_t
            #pragma unroll
            for (int r = 0; r < 8; ++r) {
                __bf16 hb = f2bf(h1v[r]);
                int m = r + 8*hi;
                h1cat[m*K1P + wave*16 + n16] = hb;
                h2cat[m*K2P + wave*16 + n16] = hb;
            }
        }
        __syncthreads();                       // (3) h1_t visible

        // ---- layer 2: z2 = [h1_t|h2] @ Uc2 ; all 8 waves, K=224 ----
        {
            const __bf16* arow = h2cat + n16*K2P;
            v16bf A2[7];
            #pragma unroll
            for (int kk = 0; kk < 7; ++kk) A2[kk] = load_a(arow, kk, hi);
            v8f z2[4];
            #pragma unroll
            for (int q = 0; q < 4; ++q) {
                const __bf16* bcol = Uc2T + (q*U2 + wave*16 + n16)*K2P + hi*16;
                v8f acc = {};
                #pragma unroll
                for (int kk = 0; kk < 7; ++kk)
                    acc = wmma_bf16(A2[kk], *(const v16bf*)(bcol + kk*32), acc);
                z2[q] = acc;
            }
            #pragma unroll
            for (int r = 0; r < 8; ++r) {       // act = relu
                float iv = sigm(z2[0][r] + bia2[0]);
                float fv = sigm(z2[1][r] + bia2[1]);
                float gv = fmaxf(z2[2][r] + bia2[2], 0.0f);
                float ov = sigm(z2[3][r] + bia2[3]);
                float c  = fv*c2[r] + iv*gv;
                c2[r] = c;
                h2v[r] = ov * fmaxf(c, 0.0f);
            }
        }
        __syncthreads();                       // (4) layer2 A-reads done

        #pragma unroll
        for (int r = 0; r < 8; ++r)            // publish h2_t
            h2cat[(r + 8*hi)*K2P + U1 + wave*16 + n16] = f2bf(h2v[r]);
    }

    // ======================== dense head ========================
    #pragma unroll
    for (int r = 0; r < 8; ++r)
        dbuf0[(r + 8*hi)*U2 + wave*16 + n16] = h2v[r];   // h2_T as f32
    __syncthreads();

    dense_layer(dbuf0, dbuf1, Wd1, bd1, 128, 80, true,  tid); __syncthreads();
    dense_layer(dbuf1, dbuf0, Wd2, bd2,  80, 64, true,  tid); __syncthreads();
    dense_layer(dbuf0, dbuf1, Wd3, bd3,  64, 32, true,  tid); __syncthreads();
    dense_layer(dbuf1, dbuf0, Wd4, bd4,  32, 10, false, tid); __syncthreads();
    dense_layer(dbuf0, dbuf1, Wd5, bd5,  10,  4, false, tid); __syncthreads();

    if (tid < BT*4)
        out[(size_t)(b0 + (tid >> 2))*4 + (tid & 3)] = dbuf1[tid];
}

extern "C" void kernel_launch(void* const* d_in, const int* in_sizes, int n_in,
                              void* d_out, int out_size, void* d_ws, size_t ws_size,
                              hipStream_t stream) {
    (void)in_sizes; (void)n_in; (void)out_size; (void)d_ws; (void)ws_size;
    const float* x   = (const float*)d_in[0];
    const float* W1  = (const float*)d_in[1];
    const float* Ur1 = (const float*)d_in[2];
    const float* b1  = (const float*)d_in[3];
    const float* W2  = (const float*)d_in[4];
    const float* Ur2 = (const float*)d_in[5];
    const float* b2  = (const float*)d_in[6];
    const float* Wd1 = (const float*)d_in[7];
    const float* bd1 = (const float*)d_in[8];
    const float* Wd2 = (const float*)d_in[9];
    const float* bd2 = (const float*)d_in[10];
    const float* Wd3 = (const float*)d_in[11];
    const float* bd3 = (const float*)d_in[12];
    const float* Wd4 = (const float*)d_in[13];
    const float* bd4 = (const float*)d_in[14];
    const float* Wd5 = (const float*)d_in[15];
    const float* bd5 = (const float*)d_in[16];
    float* out = (float*)d_out;

    dim3 grid(BATCH / BT), block(TPB);
    hipLaunchKernelGGL(lstm_fused_kernel, grid, block, SMEM_BYTES, stream,
                       x, W1, Ur1, b1, W2, Ur2, b2,
                       Wd1, bd1, Wd2, bd2, Wd3, bd3, Wd4, bd4, Wd5, bd5, out);
}
